// SambaYDecoderLayer_43980465111723
// MI455X (gfx1250) — matrix operand order repaired and checked by reading, not compiled, not run on hardware
//
#include <hip/hip_runtime.h>
#include <stdint.h>

// ---------------- problem constants ----------------
enum {
  kDModel  = 2048,
  kDState  = 16,
  kDConv   = 4,
  kDInner  = 4096,
  kHeadDim = 64,
  kNHeads  = 64,
  kConvDim = kDInner + 2 * kDState,            // 4128
  kInDim   = kDInner + kConvDim + kNHeads,     // 8288
  kBatch   = 2,
  kSeq     = 4096,
  kMRows   = kBatch * kSeq,                    // 8192
};

// ---------------- feature detection ----------------
#ifndef USE_TDM
#ifdef __has_builtin
#if __has_builtin(__builtin_amdgcn_tensor_load_to_lds) && \
    __has_builtin(__builtin_amdgcn_s_wait_tensorcnt)
#define USE_TDM 1
#endif
#endif
#endif
#ifndef USE_TDM
#define USE_TDM 0
#endif

typedef unsigned int u32x4 __attribute__((ext_vector_type(4)));
typedef int          i32x4 __attribute__((ext_vector_type(4)));
typedef int          i32x8 __attribute__((ext_vector_type(8)));
typedef __bf16       v16bf __attribute__((ext_vector_type(16)));
typedef float        v8f   __attribute__((ext_vector_type(8)));

union FragU { v16bf v; uint4 q[2]; };

// ---------------- device helpers ----------------
__device__ __forceinline__ float sigmoidf_(float x) { return 1.0f / (1.0f + __expf(-x)); }
__device__ __forceinline__ float siluf_(float x)    { return x * sigmoidf_(x); }
__device__ __forceinline__ float softplusf_(float x){ return (x > 20.f) ? x : log1pf(__expf(x)); }

__device__ __forceinline__ unsigned short f32_to_bf16(float f) {
  union { float f; unsigned u; } c; c.f = f;
  unsigned u = c.u;
  u += 0x7FFFu + ((u >> 16) & 1u);   // round-to-nearest-even
  return (unsigned short)(u >> 16);
}

// The TDM builtin carries the LDS destination only as an integer field inside
// the D#, so the compiler cannot see the LDS writes. Escape the LDS buffer
// into an asm with a memory clobber (input-only operand: the pointer VALUE is
// not laundered, so subsequent accesses keep addrspace(3) and lower to
// ds_load instead of flat_load). The escape forces the compiler to assume the
// asm (standing in for the TDM) wrote the buffer.
__device__ __forceinline__ void lds_publish(const void* p) {
  asm volatile("" : : "v"(p) : "memory");
}

// Issue one 2D tile DMA (global -> LDS) on the Tensor Data Mover.
// D# layout per CDNA5 ISA §8: group0 = count/lds_addr/global_addr/type,
// group1 = data_size, tensor dims, tile dims, strides, LDS padding.
// This toolchain's builtin takes 6 args: (g0, g1, g2, g3, g4, cpol).
__device__ __forceinline__ void tdm_load_2d(
    unsigned lds_byte_off, const void* tile_base,
    unsigned tile_w, unsigned tile_h,
    unsigned tensor_w, unsigned tensor_h,
    unsigned row_stride_elems, unsigned data_size_code,
    unsigned pad_en, unsigned pad_interval, unsigned pad_amount)
{
#if USE_TDM
  unsigned long long ga = (unsigned long long)(uintptr_t)tile_base;
  u32x4 g0;
  g0[0] = 1u;                                           // count=1, user descriptor
  g0[1] = lds_byte_off;                                 // lds_addr
  g0[2] = (unsigned)(ga & 0xFFFFFFFFu);                 // global_addr[31:0]
  g0[3] = (unsigned)((ga >> 32) & 0x1FFFFFFu) | (2u << 30); // global_addr[56:32] | type=2

  unsigned dw0 = (data_size_code & 3u) << 16;           // data_size
  if (pad_en) dw0 |= (1u << 20) | ((pad_interval & 7u) << 22) | ((pad_amount & 127u) << 25);
  i32x8 g1;
  g1[0] = (int)dw0;
  g1[1] = (int)((tensor_w & 0xFFFFu) << 16);            // tensor_dim0[15:0] @ bit48
  g1[2] = (int)((tensor_w >> 16) | ((tensor_h & 0xFFFFu) << 16)); // dim0 hi | dim1 lo
  g1[3] = (int)((tensor_h >> 16) | ((tile_w & 0xFFFFu) << 16));   // dim1 hi | tile_dim0
  g1[4] = (int)(tile_h & 0xFFFFu);                      // tile_dim1 (tile_dim2 = 0)
  g1[5] = (int)row_stride_elems;                        // tensor_dim0_stride[31:0]
  g1[6] = 0;                                            // stride hi | dim1_stride lo
  g1[7] = 0;
  i32x4 z4 = {0, 0, 0, 0};
  i32x8 z8 = {0, 0, 0, 0, 0, 0, 0, 0};
  __builtin_amdgcn_tensor_load_to_lds(g0, g1, z4, z4, z8, 0);
#else
  (void)lds_byte_off; (void)tile_base; (void)tile_w; (void)tile_h;
  (void)tensor_w; (void)tensor_h; (void)row_stride_elems; (void)data_size_code;
  (void)pad_en; (void)pad_interval; (void)pad_amount;
#endif
}

#if USE_TDM
#define WAIT_TENSORCNT(n) __builtin_amdgcn_s_wait_tensorcnt(n)
#else
#define WAIT_TENSORCNT(n) do {} while (0)
#endif

// ---------------- bf16 conversion ----------------
__global__ void f32_to_bf16_kernel(const float* __restrict__ in,
                                   unsigned short* __restrict__ out, size_t n) {
  size_t i = (size_t)blockIdx.x * blockDim.x + threadIdx.x;
  if (i < n) out[i] = f32_to_bf16(in[i]);
}

// ---------------- WMMA GEMM: C[M,N] = A[M,K] * B[K,N] (bf16 in, f32 out) ----
// Block: 256 threads (8 waves). Tile 128x128, K-step 32 (one v_wmma per tile).
// LDS tiles are DMA'd by the TDM with +16B/row padding:
//   A tile: 128 rows x 32 bf16  -> row stride 80B
//   B tile: 32 rows x 128 bf16  -> row stride 272B
__global__ __launch_bounds__(256, 1)
void gemm_bf16_tdm_wmma(const unsigned short* __restrict__ A,
                        const unsigned short* __restrict__ B,
                        float* __restrict__ C, int M, int N, int K)
{
  constexpr int A_STRIDE = 80, B_STRIDE = 272;
  constexpr int A_BYTES = 128 * A_STRIDE;   // 10240
  constexpr int B_BYTES = 32 * B_STRIDE;    // 8704
  __shared__ char smem[2 * A_BYTES + 2 * B_BYTES];

  const int tid   = threadIdx.x;
  const int lane  = tid & 31;
  const int wid   = tid >> 5;
  const int waveM = wid & 3;                // 0..3 -> 32-row strip
  const int waveN = wid >> 2;               // 0..1 -> 64-col strip
  const int rowBase = blockIdx.y * 128;
  const int colBase = blockIdx.x * 128;
  const int r15  = lane & 15;
  const int half = lane >> 4;

  v8f zero8 = {0.f, 0.f, 0.f, 0.f, 0.f, 0.f, 0.f, 0.f};
  v8f acc[2][4];
  for (int mt = 0; mt < 2; ++mt)
    for (int nt = 0; nt < 4; ++nt) acc[mt][nt] = zero8;

  const int KT = K / 32;

#if USE_TDM
  if (wid == 0) {   // wave 0 drives the TDM; EXEC-independent, one issue per wave
    tdm_load_2d(0u, A + (size_t)rowBase * K, 32, 128,
                (unsigned)K, (unsigned)(M - rowBase), (unsigned)K, 1, 1, 3, 3);
    tdm_load_2d((unsigned)(2 * A_BYTES), B + colBase, 128, 32,
                (unsigned)(N - colBase), (unsigned)K, (unsigned)N, 1, 1, 5, 3);
  }
#endif

  for (int kt = 0; kt < KT; ++kt) {
    const int buf = kt & 1;
    const int k0  = kt * 32;
    const unsigned aoff  = buf ? (unsigned)A_BYTES : 0u;
    const unsigned boff  = (unsigned)(2 * A_BYTES) + (buf ? (unsigned)B_BYTES : 0u);

#if USE_TDM
    if (wid == 0) {
      if (kt + 1 < KT) {
        const int k1 = k0 + 32;
        const unsigned aoffN = buf ? 0u : (unsigned)A_BYTES;
        const unsigned boffN = (unsigned)(2 * A_BYTES) + (buf ? 0u : (unsigned)B_BYTES);
        // issue next tile into the other buffer, then wait for current (<=2 in flight)
        tdm_load_2d(aoffN, A + (size_t)rowBase * K + k1, 32, 128,
                    (unsigned)(K - k1), (unsigned)(M - rowBase), (unsigned)K, 1, 1, 3, 3);
        tdm_load_2d(boffN, B + (size_t)k1 * N + colBase, 128, 32,
                    (unsigned)(N - colBase), (unsigned)(K - k1), (unsigned)N, 1, 1, 5, 3);
        WAIT_TENSORCNT(2);
      } else {
        WAIT_TENSORCNT(0);
      }
    }
#else
    // Fallback: cooperative synchronous tile load into the same padded layout.
    for (int i = tid; i < 128 * 32; i += 256) {
      int r = i >> 5, c = i & 31;
      *(unsigned short*)(smem + aoff + r * A_STRIDE + c * 2) =
          A[(size_t)(rowBase + r) * K + k0 + c];
    }
    for (int i = tid; i < 32 * 128; i += 256) {
      int r = i >> 7, c = i & 127;
      int col = colBase + c;
      *(unsigned short*)(smem + boff + r * B_STRIDE + c * 2) =
          (col < N) ? B[(size_t)(k0 + r) * N + col] : (unsigned short)0;
    }
#endif
    __syncthreads();

    // Publish the DMA'd LDS contents to the compiler (see lds_publish).
    lds_publish(smem);
    const char* sA = smem + aoff;
    const char* sB = smem + boff;

    // A fragments per documented 16-bit 16x32 layout:
    // lanes 0-15: K 0..7 then 16..23; lanes 16-31: K 8..15 then 24..31.
    FragU af[2];
#pragma unroll
    for (int mt = 0; mt < 2; ++mt) {
      const char* pa = sA + (waveM * 32 + mt * 16 + r15) * A_STRIDE + half * 16;
      af[mt].q[0] = *(const uint4*)pa;
      af[mt].q[1] = *(const uint4*)(pa + 32);
    }
    // B fragments: lane%16 = N column; K in same order as A (8h+s, then 16+8h+s).
    FragU bfr[4];
#pragma unroll
    for (int nt = 0; nt < 4; ++nt) {
      const char* pb = sB + (waveN * 64 + nt * 16 + r15) * 2;
#pragma unroll
      for (int s = 0; s < 16; ++s) {
        const int kk = (s < 8) ? (half * 8 + s) : (16 + half * 8 + (s - 8));
        bfr[nt].v[s] = *(const __bf16*)(pb + kk * B_STRIDE);
      }
    }
#pragma unroll
    for (int mt = 0; mt < 2; ++mt)
#pragma unroll
      for (int nt = 0; nt < 4; ++nt)
        acc[mt][nt] = __builtin_amdgcn_wmma_f32_16x16x32_bf16(
            false, af[mt].v, false, bfr[nt].v, (short)0, acc[mt][nt], false, false);

    __syncthreads();
  }

  // C/D layout: lanes0-15 {M=r,N=lane}, lanes16-31 {M=8+r,N=lane-16}.
#pragma unroll
  for (int mt = 0; mt < 2; ++mt) {
    const int row0 = rowBase + waveM * 32 + mt * 16 + half * 8;
#pragma unroll
    for (int nt = 0; nt < 4; ++nt) {
      const int col = colBase + waveN * 64 + nt * 16 + r15;
      if (col < N) {
#pragma unroll
        for (int r = 0; r < 8; ++r)
          C[(size_t)(row0 + r) * N + col] = acc[mt][nt][r];
      }
    }
  }
}

// ---------------- depthwise causal conv(4) + SiLU --------------------------
__global__ void conv_silu_kernel(const float* __restrict__ zx,
                                 const float* __restrict__ cw,
                                 const float* __restrict__ cb,
                                 float* __restrict__ out)
{
  size_t idx = (size_t)blockIdx.x * blockDim.x + threadIdx.x;
  const size_t total = (size_t)kMRows * kConvDim;
  if (idx >= total) return;
  int c = (int)(idx % kConvDim);
  int m = (int)(idx / kConvDim);
  int l = m % kSeq;
  float accv = cb[c];
#pragma unroll
  for (int k = 0; k < kDConv; ++k) {
    int lt = l + k - (kDConv - 1);
    if (lt >= 0)
      accv += zx[(size_t)(m + (lt - l)) * kInDim + kDInner + c] * cw[c * kDConv + k];
  }
  out[idx] = siluf_(accv);
}

// ---------------- dt = softplus(raw + bias), dA = exp(dt * -exp(A_log)) ----
__global__ void dt_dA_kernel(const float* __restrict__ zx,
                             const float* __restrict__ dt_bias,
                             const float* __restrict__ A_log,
                             float* __restrict__ dt, float* __restrict__ dA)
{
  int idx = blockIdx.x * blockDim.x + threadIdx.x;
  if (idx >= kMRows * kNHeads) return;
  int h = idx % kNHeads;
  int m = idx / kNHeads;
  float raw = zx[(size_t)m * kInDim + kDInner + kConvDim + h] + dt_bias[h];
  float d = softplusf_(raw);
  float a = -__expf(A_log[h]);
  dt[idx] = d;
  dA[idx] = __expf(d * a);
}

// ---------------- sequential selective-state scan --------------------------
// One block per (batch, head). 128 threads; thread (p = tid/2) owns 8 of the
// 16 state columns of row p. Inputs staged in 64-step LDS chunks.
__global__ __launch_bounds__(128, 1)
void ssm_scan_kernel(const float* __restrict__ convout,
                     const float* __restrict__ dtb,
                     const float* __restrict__ dAb,
                     const float* __restrict__ Dk_,
                     float* __restrict__ y)
{
  const int bh = blockIdx.x;
  const int b = bh / kNHeads, h = bh % kNHeads;
  const int tid = threadIdx.x;
  const int p  = tid >> 1;
  const int nb = (tid & 1) * 8;

  __shared__ float xch[64][kHeadDim];
  __shared__ float Bch[64][kDState];
  __shared__ float Cch[64][kDState];
  __shared__ float dtch[64], dAch[64];

  float hreg[8];
#pragma unroll
  for (int j = 0; j < 8; ++j) hreg[j] = 0.f;
  const float Dk = Dk_[h];

  for (int c0 = 0; c0 < kSeq; c0 += 64) {
    for (int i = tid; i < 64 * kHeadDim; i += 128) {
      int t = i >> 6, d = i & 63;
      size_t m = (size_t)(b * kSeq + c0 + t);
      xch[t][d] = convout[m * kConvDim + h * kHeadDim + d];
    }
    for (int i = tid; i < 64 * kDState; i += 128) {
      int t = i >> 4, n = i & 15;
      size_t m = (size_t)(b * kSeq + c0 + t);
      Bch[t][n] = convout[m * kConvDim + kDInner + n];
      Cch[t][n] = convout[m * kConvDim + kDInner + kDState + n];
    }
    if (tid < 64) {
      size_t m = (size_t)(b * kSeq + c0 + tid);
      dtch[tid] = dtb[m * kNHeads + h];
      dAch[tid] = dAb[m * kNHeads + h];
    }
    __syncthreads();

    for (int t = 0; t < 64; ++t) {
      float dA_t = dAch[t], dt_t = dtch[t];
      float xv = xch[t][p];
      float coef = dt_t * xv;
      float sum = 0.f;
#pragma unroll
      for (int j = 0; j < 8; ++j) {
        int n = nb + j;
        hreg[j] = hreg[j] * dA_t + coef * Bch[t][n];
        sum += hreg[j] * Cch[t][n];
      }
      sum += __shfl_xor(sum, 1, 32);   // pair-reduce within wave32
      if ((tid & 1) == 0) {
        size_t m = (size_t)(b * kSeq + c0 + t);
        y[m * (size_t)kDInner + h * kHeadDim + p] = sum + Dk * xv;
      }
    }
    __syncthreads();
  }
}

// ---------------- gate*silu + RMSNorm -> bf16 ------------------------------
__global__ __launch_bounds__(256, 1)
void gate_rmsnorm_kernel(const float* __restrict__ y,
                         const float* __restrict__ zx,
                         const float* __restrict__ norm_w,
                         unsigned short* __restrict__ ybf)
{
  const int m = blockIdx.x, tid = threadIdx.x;
  float g[16];
  float ss = 0.f;
#pragma unroll
  for (int j = 0; j < 16; ++j) {
    int i = tid + j * 256;
    float gv = zx[(size_t)m * kInDim + i];
    float v = y[(size_t)m * kDInner + i] * siluf_(gv);
    g[j] = v;
    ss += v * v;
  }
  __shared__ float red[256];
  red[tid] = ss;
  __syncthreads();
  for (int s = 128; s > 0; s >>= 1) {
    if (tid < s) red[tid] += red[tid + s];
    __syncthreads();
  }
  float scale = rsqrtf(red[0] * (1.0f / kDInner) + 1e-5f);
#pragma unroll
  for (int j = 0; j < 16; ++j) {
    int i = tid + j * 256;
    ybf[(size_t)m * kDInner + i] = f32_to_bf16(g[j] * scale * norm_w[i]);
  }
}

// ---------------- launcher -------------------------------------------------
extern "C" void kernel_launch(void* const* d_in, const int* in_sizes, int n_in,
                              void* d_out, int out_size, void* d_ws, size_t ws_size,
                              hipStream_t stream)
{
  (void)in_sizes; (void)n_in; (void)out_size; (void)ws_size;
  const float* hidden  = (const float*)d_in[0];
  const float* W_in    = (const float*)d_in[1];
  const float* conv_w  = (const float*)d_in[2];
  const float* conv_b  = (const float*)d_in[3];
  const float* A_log   = (const float*)d_in[4];
  const float* D_skip  = (const float*)d_in[5];
  const float* dt_bias = (const float*)d_in[6];
  const float* norm_w  = (const float*)d_in[7];
  const float* W_out   = (const float*)d_in[8];
  float* out = (float*)d_out;

  char* ws = (char*)d_ws;
  size_t off = 0;
  auto take = [&](size_t bytes) -> void* {
    void* p = (void*)(ws + off);
    off += (bytes + 255) & ~(size_t)255;
    return p;
  };
  unsigned short* Xbf    = (unsigned short*)take((size_t)kMRows * kDModel * 2);
  unsigned short* Winbf  = (unsigned short*)take((size_t)kDModel * kInDim * 2);
  unsigned short* Woutbf = (unsigned short*)take((size_t)kDInner * kDModel * 2);
  float* zx    = (float*)take((size_t)kMRows * kInDim * 4);
  float* convo = (float*)take((size_t)kMRows * kConvDim * 4);
  float* dtv   = (float*)take((size_t)kMRows * kNHeads * 4);
  float* dAv   = (float*)take((size_t)kMRows * kNHeads * 4);
  float* ybuf  = (float*)take((size_t)kMRows * kDInner * 4);
  unsigned short* ybf = (unsigned short*)take((size_t)kMRows * kDInner * 2);

  // 1) bf16 conversions
  {
    size_t n1 = (size_t)kMRows * kDModel;
    f32_to_bf16_kernel<<<(unsigned)((n1 + 255) / 256), 256, 0, stream>>>(hidden, Xbf, n1);
    size_t n2 = (size_t)kDModel * kInDim;
    f32_to_bf16_kernel<<<(unsigned)((n2 + 255) / 256), 256, 0, stream>>>(W_in, Winbf, n2);
    size_t n3 = (size_t)kDInner * kDModel;
    f32_to_bf16_kernel<<<(unsigned)((n3 + 255) / 256), 256, 0, stream>>>(W_out, Woutbf, n3);
  }
  // 2) input projection GEMM: [8192 x 2048] @ [2048 x 8288]
  {
    dim3 g((kInDim + 127) / 128, kMRows / 128);
    gemm_bf16_tdm_wmma<<<g, 256, 0, stream>>>(Xbf, Winbf, zx, kMRows, kInDim, kDModel);
  }
  // 3) conv + silu
  {
    size_t total = (size_t)kMRows * kConvDim;
    conv_silu_kernel<<<(unsigned)((total + 255) / 256), 256, 0, stream>>>(zx, conv_w, conv_b, convo);
  }
  // 4) dt / dA
  dt_dA_kernel<<<(kMRows * kNHeads + 255) / 256, 256, 0, stream>>>(zx, dt_bias, A_log, dtv, dAv);
  // 5) sequential scan: one block per (batch, head)
  ssm_scan_kernel<<<kBatch * kNHeads, 128, 0, stream>>>(convo, dtv, dAv, D_skip, ybuf);
  // 6) gated RMSNorm -> bf16
  gate_rmsnorm_kernel<<<kMRows, 256, 0, stream>>>(ybuf, zx, norm_w, ybf);
  // 7) output projection GEMM: [8192 x 4096] @ [4096 x 2048]
  {
    dim3 g(kDModel / 128, kMRows / 128);
    gemm_bf16_tdm_wmma<<<g, 256, 0, stream>>>(ybf, Woutbf, out, kMRows, kDModel, kDInner);
  }
}
